// Model_26869315404370
// MI455X (gfx1250) — matrix operand order repaired and checked by reading, not compiled, not run on hardware
//
#include <hip/hip_runtime.h>
#include <hip/hip_bf16.h>
#include <math.h>

// ---------------- types for WMMA (gfx1250, wave32) ----------------
typedef __attribute__((ext_vector_type(16))) __bf16 v16bf;
typedef __attribute__((ext_vector_type(8)))  float  v8f;

#define B_    64
#define S_    512
#define E_    256
#define H_    512
#define G_    2048          // 4*H
#define T_    31
#define MROWS (S_ * B_)     // 32768 rows, row = s*64 + b
#define MT_   (MROWS / 16)  // 2048
#define KT_E  (E_ / 32)     // 8
#define KT_H  (H_ / 32)     // 16

// ---------------- workspace layout (bytes) ----------------
#define OFF_XFRAG  0ull
#define SZ_XFRAG   ((size_t)MROWS * E_ * 2)        // 16 MB  bf16 A-fragments of X
#define OFF_WIHF   (OFF_XFRAG + SZ_XFRAG)
#define SZ_WIH     ((size_t)G_ * E_ * 2)           // 1 MB
#define OFF_WIHB   (OFF_WIHF + SZ_WIH)
#define OFF_WHHF   (OFF_WIHB + SZ_WIH)
#define SZ_WHH     ((size_t)G_ * H_ * 2)           // 2 MB
#define OFF_WHHB   (OFF_WHHF + SZ_WHH)
#define OFF_XGF    (OFF_WHHB + SZ_WHH)
#define SZ_XG      ((size_t)MROWS * G_ * 4)        // 256 MB f32 gate pre-activations
#define OFF_XGB    (OFF_XGF + SZ_XG)
#define OFF_HF     (OFF_XGB + SZ_XG)
#define SZ_H       ((size_t)MROWS * H_ * 4)        // 64 MB  f32 hidden states [s][b][h]
#define OFF_HB     (OFF_HF + SZ_H)
#define OFF_EM     (OFF_HB + SZ_H)
#define SZ_EM      ((size_t)B_ * S_ * T_ * 4)      // emissions [b][s][t]
#define OFF_BP     (OFF_EM + SZ_EM)                // backpointers [t-1][b][t] int32

// ---------------- fragment index helpers (ISA 7.12.2 layouts) ----------------
// 16-bit A-matrix 16x32: lanes 0-15 hold K in {0..7,16..23}, lanes 16-31 hold {8..15,24..31}
__device__ __forceinline__ void a_frag_pos(int mm, int kk, int& lane, int& el) {
    int group = kk >> 3;                       // 0..3
    lane = mm + ((group & 1) ? 16 : 0);
    el   = kk - (((group + 1) >> 1) << 3);     // g0:kk g1:kk-8 g2:kk-8 g3:kk-16
}
// 16-bit B-matrix 32x16: lanes 0-15 hold K in {0..7,16..23}, lanes 16-31 hold {8..15,24..31}
__device__ __forceinline__ void b_frag_pos(int kk, int nn, int& lane, int& el) {
    int group = kk >> 3;
    lane = nn + ((group & 1) ? 16 : 0);
    el   = kk - (((group + 1) >> 1) << 3);
}

__device__ __forceinline__ v16bf load_frag(const __bf16* base, int fragIdx, int lane) {
    return *(const v16bf*)(base + ((size_t)fragIdx * 32 + lane) * 16);
}
__device__ __forceinline__ v8f wmma_bf16(v16bf a, v16bf b, v8f c) {
    return __builtin_amdgcn_wmma_f32_16x16x32_bf16(false, a, false, b, (short)0, c, false, false);
}
__device__ __forceinline__ float sigf(float x)  { return 1.0f / (1.0f + __expf(-x)); }
__device__ __forceinline__ float tanhf_(float x){ return 2.0f / (1.0f + __expf(-2.0f * x)) - 1.0f; }

// ---------------- 1) embedding gather + pack X into A-fragments ----------------
__global__ void k_gather_pack_x(const int* __restrict__ ids,
                                const float* __restrict__ embed,
                                __bf16* __restrict__ xfrag) {
    int idx = blockIdx.x * blockDim.x + threadIdx.x;     // MROWS*E_ threads
    int row = idx >> 8;            // s*64 + b
    int e   = idx & 255;
    int s = row >> 6, b = row & 63;
    int id = ids[b * S_ + s];
    float v = embed[(size_t)id * E_ + e];
    int mt = row >> 4, mm = row & 15;
    int kt = e >> 5,  kk = e & 31;
    int lane, el; a_frag_pos(mm, kk, lane, el);
    xfrag[(((size_t)mt * KT_E + kt) * 32 + lane) * 16 + el] = (__bf16)v;
}

// ---------------- 2) pack W (G x K, row-major) as B = W^T fragments ----------------
__global__ void k_pack_w(const float* __restrict__ w, __bf16* __restrict__ dst, int K) {
    int idx = blockIdx.x * blockDim.x + threadIdx.x;     // G_*K threads
    int n = idx / K;               // gate/output column
    int k = idx - n * K;           // reduction index
    int nt = n >> 4, nn = n & 15;
    int kt = k >> 5, kk = k & 31;
    int lane, el; b_frag_pos(kk, nn, lane, el);
    int KT = K >> 5;
    dst[(((size_t)nt * KT + kt) * 32 + lane) * 16 + el] = (__bf16)w[(size_t)n * K + k];
}

// ---------------- 3) input-projection GEMM: xg = X @ W_ih^T + b ----------------
// wave-per-(dir, mtile, 4-ntile-group).  2 * 2048 * 32 = 131072 waves.
// Software-pipelined: next K-step's fragments are in flight while current WMMAs issue.
__global__ void k_xg_gemm(const __bf16* __restrict__ xfrag,
                          const __bf16* __restrict__ wF, const __bf16* __restrict__ wB,
                          const float* __restrict__ bF,  const float* __restrict__ bB,
                          float* __restrict__ xgF, float* __restrict__ xgB) {
    int gw   = (blockIdx.x * blockDim.x + threadIdx.x) >> 5;
    int lane = threadIdx.x & 31;
    int dir  = gw >> 16;                 // 65536 waves per direction
    int rem  = gw & 65535;
    int mt   = rem >> 5;                 // 0..2047
    int ntg  = rem & 31;                 // 0..31 -> 4 nTiles each
    const __bf16* wf  = dir ? wB : wF;
    const float*  bia = dir ? bB : bF;
    float*        xg  = dir ? xgB : xgF;
    int nb = ntg * 4;

    v8f acc0 = {}, acc1 = {}, acc2 = {}, acc3 = {};
    v16bf a  = load_frag(xfrag, mt * KT_E, lane);
    v16bf b0 = load_frag(wf, (nb + 0) * KT_E, lane);
    v16bf b1 = load_frag(wf, (nb + 1) * KT_E, lane);
    v16bf b2 = load_frag(wf, (nb + 2) * KT_E, lane);
    v16bf b3 = load_frag(wf, (nb + 3) * KT_E, lane);
#pragma unroll
    for (int kt = 0; kt < KT_E; ++kt) {
        v16bf an = a, c0 = b0, c1 = b1, c2 = b2, c3 = b3;
        if (kt + 1 < KT_E) {             // issue next K-step's loads before consuming
            an = load_frag(xfrag, mt * KT_E + kt + 1, lane);
            c0 = load_frag(wf, (nb + 0) * KT_E + kt + 1, lane);
            c1 = load_frag(wf, (nb + 1) * KT_E + kt + 1, lane);
            c2 = load_frag(wf, (nb + 2) * KT_E + kt + 1, lane);
            c3 = load_frag(wf, (nb + 3) * KT_E + kt + 1, lane);
        }
        acc0 = wmma_bf16(a, b0, acc0);
        acc1 = wmma_bf16(a, b1, acc1);
        acc2 = wmma_bf16(a, b2, acc2);
        acc3 = wmma_bf16(a, b3, acc3);
        a = an; b0 = c0; b1 = c1; b2 = c2; b3 = c3;
    }
    // C/D layout: vgpr i, lane l -> M = i + (l>=16?8:0), N = l&15
    int rowOff = (lane >= 16) ? 8 : 0;
    int nn = lane & 15;
    v8f accs[4] = {acc0, acc1, acc2, acc3};
    for (int j = 0; j < 4; ++j) {
        int n0 = (ntg * 4 + j) * 16 + nn;
        float bv = bia[n0];
        for (int i = 0; i < 8; ++i) {
            int m = mt * 16 + i + rowOff;
            xg[(size_t)m * G_ + n0] = accs[j][i] + bv;
        }
    }
}

// ---------------- 4) persistent LSTM recurrence (one WGP per direction) ----------------
// LDS: hfrag double buffer (2 x 64KB bf16 A-fragments of h) + f32 cell state (128KB)
__global__ void __launch_bounds__(1024, 1)
k_lstm(const float* __restrict__ xgF, const float* __restrict__ xgB,
       const __bf16* __restrict__ whhF, const __bf16* __restrict__ whhB,
       float* __restrict__ hF, float* __restrict__ hB) {
    extern __shared__ char smem[];
    __bf16* hfrag0 = (__bf16*)smem;                 // 4*16 frags * 512 = 32768 elems
    __bf16* hfrag1 = hfrag0 + 32768;
    float*  cst    = (float*)(smem + 131072);       // [64][512]

    const int dir  = blockIdx.x;
    const int tid  = threadIdx.x;
    const int wave = tid >> 5, lane = tid & 31;
    const __bf16* wf = dir ? whhB : whhF;
    const float*  xg = dir ? xgB  : xgF;
    float*        ho = dir ? hB   : hF;

    for (int i = tid; i < 32768; i += 1024) { hfrag0[i] = (__bf16)0.0f; cst[i] = 0.0f; }
    __syncthreads();

    const int rowOff = (lane >= 16) ? 8 : 0;
    const int nn = lane & 15;

    for (int t = 0; t < S_; ++t) {
        int s = dir ? (S_ - 1 - t) : t;
        __bf16* hcur = (t & 1) ? hfrag1 : hfrag0;
        __bf16* hnxt = (t & 1) ? hfrag0 : hfrag1;
        size_t xbase = (size_t)s * B_ * G_;

#pragma unroll 1
        for (int q = 0; q < 4; ++q) {
            int task = wave * 4 + q;        // 128 tasks: 4 batch-tiles x 32 hidden blocks
            int mt   = task & 3;
            int jblk = task >> 2;           // hidden cols [jblk*16, jblk*16+16)
            v8f aI = {}, aF = {}, aG = {}, aO = {};
            // software pipeline: A from LDS, 4 gate B-fragments from L2
            v16bf a  = *(const v16bf*)(hcur + (((size_t)(mt * KT_H)) * 32 + lane) * 16);
            v16bf bi = load_frag(wf, (jblk      ) * KT_H, lane);
            v16bf bf = load_frag(wf, (jblk +  32) * KT_H, lane);
            v16bf bg = load_frag(wf, (jblk +  64) * KT_H, lane);
            v16bf bo = load_frag(wf, (jblk +  96) * KT_H, lane);
#pragma unroll
            for (int kt = 0; kt < KT_H; ++kt) {
                v16bf an = a, ci = bi, cf = bf, cg = bg, co = bo;
                if (kt + 1 < KT_H) {
                    an = *(const v16bf*)(hcur + (((size_t)(mt * KT_H + kt + 1)) * 32 + lane) * 16);
                    ci = load_frag(wf, (jblk      ) * KT_H + kt + 1, lane);
                    cf = load_frag(wf, (jblk +  32) * KT_H + kt + 1, lane);
                    cg = load_frag(wf, (jblk +  64) * KT_H + kt + 1, lane);
                    co = load_frag(wf, (jblk +  96) * KT_H + kt + 1, lane);
                }
                aI = wmma_bf16(a, bi, aI);
                aF = wmma_bf16(a, bf, aF);
                aG = wmma_bf16(a, bg, aG);
                aO = wmma_bf16(a, bo, aO);
                a = an; bi = ci; bf = cf; bg = cg; bo = co;
            }
            int j = jblk * 16 + nn;         // hidden unit index
            int kt2 = j >> 5, kk = j & 31;
            int lane2, el; a_frag_pos(0, kk, lane2, el);   // mm added below
            for (int i = 0; i < 8; ++i) {
                int m = mt * 16 + i + rowOff;               // batch index
                size_t xr = xbase + (size_t)m * G_;
                float gi = aI[i] + xg[xr + j];
                float gf = aF[i] + xg[xr + H_ + j];
                float gg = aG[i] + xg[xr + 2 * H_ + j];
                float go = aO[i] + xg[xr + 3 * H_ + j];
                float cp = cst[m * H_ + j];
                float c  = sigf(gf) * cp + sigf(gi) * tanhf_(gg);
                float h  = sigf(go) * tanhf_(c);
                cst[m * H_ + j] = c;
                ho[((size_t)s * B_ + m) * H_ + j] = h;
                hnxt[(((size_t)(mt * KT_H + kt2)) * 32 + (lane2 + (m & 15))) * 16 + el] = (__bf16)h;
            }
        }
        __syncthreads();   // hnxt complete before next step reads it
    }
}

// ---------------- 5) emissions: em = [hf|hb] @ lin_w^T + lin_b ----------------
__global__ void k_emissions(const float* __restrict__ hF, const float* __restrict__ hB,
                            const float* __restrict__ lw, const float* __restrict__ lb,
                            float* __restrict__ em) {
    __shared__ float sh[2 * H_];
    int bs = blockIdx.x;           // b*512 + s
    int b = bs >> 9, s = bs & 511;
    int tid = threadIdx.x;         // 32
    size_t hrow = ((size_t)s * B_ + b) * H_;
    for (int i = tid; i < H_; i += 32) { sh[i] = hF[hrow + i]; sh[H_ + i] = hB[hrow + i]; }
    __syncthreads();
    if (tid < T_) {
        float sum = lb[tid];
        const float* w = lw + (size_t)tid * (2 * H_);
        for (int k = 0; k < 2 * H_; ++k) sum += sh[k] * w[k];
        em[((size_t)b * S_ + s) * T_ + tid] = sum;
    }
}

// ---------------- 6) Viterbi: one wave per batch, lane = tag ----------------
__global__ void k_viterbi(const float* __restrict__ em, const int* __restrict__ mask,
                          const float* __restrict__ trans, const float* __restrict__ st,
                          const float* __restrict__ en, int* __restrict__ bp,
                          float* __restrict__ out) {
    int b = blockIdx.x;
    int j = threadIdx.x;           // 32 lanes, tags 0..30 active
    bool act = j < T_;
    float tr[T_];
    for (int i = 0; i < T_; ++i) tr[i] = act ? trans[i * T_ + j] : 0.0f;
    float score = act ? (st[j] + em[(size_t)b * S_ * T_ + j]) : -3.0e38f;

    for (int t = 1; t < S_; ++t) {
        float best = -3.0e38f; int bi = 0;
        for (int i = 0; i < T_; ++i) {
            float si = __shfl(score, i, 32);
            float cand = si + tr[i];
            if (cand > best) { best = cand; bi = i; }   // first-max like jnp.argmax
        }
        bool m = mask[b * S_ + t] != 0;
        float ns = best + em[((size_t)b * S_ + t) * T_ + (act ? j : 0)];
        if (act) {
            bp[(((size_t)(t - 1)) * B_ + b) * T_ + j] = m ? bi : j;
            if (m) score = ns;
        }
    }
    float fin = act ? (score + en[j]) : -3.0e38f;
    float best = -3.0e38f; int bj = 0;
    for (int i = 0; i < T_; ++i) {
        float v = __shfl(fin, i, 32);
        if (v > best) { best = v; bj = i; }
    }
    if (j == 0) {
        out[(size_t)B_ * S_ + b] = best;                 // best_score after paths
        int tag = bj;
        out[(size_t)b * S_ + (S_ - 1)] = (float)tag;
        for (int t = S_ - 2; t >= 0; --t) {
            tag = bp[(((size_t)t) * B_ + b) * T_ + tag];
            out[(size_t)b * S_ + t] = (float)tag;
        }
    }
}

// ---------------- launcher ----------------
extern "C" void kernel_launch(void* const* d_in, const int* in_sizes, int n_in,
                              void* d_out, int out_size, void* d_ws, size_t ws_size,
                              hipStream_t stream) {
    const int*   ids   = (const int*)  d_in[0];
    const int*   mask  = (const int*)  d_in[1];
    const float* embed = (const float*)d_in[2];
    const float* wihF  = (const float*)d_in[3];
    const float* whhF  = (const float*)d_in[4];
    const float* bFg   = (const float*)d_in[5];
    const float* wihB  = (const float*)d_in[6];
    const float* whhB  = (const float*)d_in[7];
    const float* bBg   = (const float*)d_in[8];
    const float* linw  = (const float*)d_in[9];
    const float* linb  = (const float*)d_in[10];
    const float* trans = (const float*)d_in[11];
    const float* strt  = (const float*)d_in[12];
    const float* endt  = (const float*)d_in[13];

    char* ws = (char*)d_ws;
    __bf16* xfrag   = (__bf16*)(ws + OFF_XFRAG);
    __bf16* wihFf   = (__bf16*)(ws + OFF_WIHF);
    __bf16* wihBf   = (__bf16*)(ws + OFF_WIHB);
    __bf16* whhFf   = (__bf16*)(ws + OFF_WHHF);
    __bf16* whhBf   = (__bf16*)(ws + OFF_WHHB);
    float*  xgF     = (float*)(ws + OFF_XGF);
    float*  xgB     = (float*)(ws + OFF_XGB);
    float*  hF      = (float*)(ws + OFF_HF);
    float*  hB      = (float*)(ws + OFF_HB);
    float*  em      = (float*)(ws + OFF_EM);
    int*    bp      = (int*)  (ws + OFF_BP);
    float*  outp    = (float*)d_out;

    // 1) embedding gather + X fragment pack: 32768*256 elements
    k_gather_pack_x<<<(MROWS * E_) / 256, 256, 0, stream>>>(ids, embed, xfrag);

    // 2) weight fragment packs
    k_pack_w<<<(G_ * E_) / 256, 256, 0, stream>>>(wihF, wihFf, E_);
    k_pack_w<<<(G_ * E_) / 256, 256, 0, stream>>>(wihB, wihBf, E_);
    k_pack_w<<<(G_ * H_) / 256, 256, 0, stream>>>(whhF, whhFf, H_);
    k_pack_w<<<(G_ * H_) / 256, 256, 0, stream>>>(whhB, whhBf, H_);

    // 3) input projection GEMM (both directions): 131072 waves
    k_xg_gemm<<<16384, 256, 0, stream>>>(xfrag, wihFf, wihBf, bFg, bBg, xgF, xgB);

    // 4) recurrent LSTM: 2 blocks (fwd/bwd), 1024 threads, 256KB dynamic LDS
    (void)hipFuncSetAttribute((const void*)k_lstm,
                              hipFuncAttributeMaxDynamicSharedMemorySize, 262144);
    k_lstm<<<2, 1024, 262144, stream>>>(xgF, xgB, whhFf, whhBf, hF, hB);

    // 5) emissions
    k_emissions<<<B_ * S_, 32, 0, stream>>>(hF, hB, linw, linb, em);

    // 6) viterbi + backtrace -> d_out (paths [64*512] then best_score [64])
    k_viterbi<<<B_, 32, 0, stream>>>(em, mask, trans, strt, endt, bp, outp);

    (void)in_sizes; (void)n_in; (void)out_size; (void)ws_size;
}
